// Head_37864431682467
// MI455X (gfx1250) — matrix operand order repaired and checked by reading, not compiled, not run on hardware
//
#include <hip/hip_runtime.h>

// Problem constants (from reference): B=8, T=2048, D=1024, Q=128
#define B_   8
#define T_   2048
#define D_   1024
#define QD_  128
#define BT_  (B_*T_)
#define NEGV (-1.0e9f)

typedef __bf16 bf16_t;
typedef bf16_t v16bf __attribute__((ext_vector_type(16)));
typedef float  v8f   __attribute__((ext_vector_type(8)));
typedef unsigned int   u32;
typedef unsigned short u16;

union ABReg { v16bf bf; u32 u[8]; uint4 q[2]; };

static __device__ inline u16 cvt_bf16(float a) {
  u32 ua = __float_as_uint(a);
  return (u16)((ua + 0x7FFFu + ((ua >> 16) & 1u)) >> 16);   // round-to-nearest-even
}
static __device__ inline u32 pack_bf16(float a, float b) {
  return (u32)cvt_bf16(a) | ((u32)cvt_bf16(b) << 16);
}

// ---- WMMA operand loaders (wave32 register layouts per CDNA5 ISA 7.12.2) ----
// A matrix 16x32 bf16: lane L (and L+16) holds row M=L&15.
//   VGPR0-3: K = hi*8 + {0..7};  VGPR4-7: K = 16 + hi*8 + {0..7}   (hi = lane>>4)
static __device__ inline v16bf load_A_bf16(const u16* base, int ld, int r16, int hi) {
  ABReg r;
  const u16* p = base + (size_t)r16 * ld + hi * 8;
  r.q[0] = *(const uint4*)(p);        // 8 bf16: K = hi*8 .. hi*8+7
  r.q[1] = *(const uint4*)(p + 16);   // 8 bf16: K = 16+hi*8 .. +7
  return r.bf;
}
static __device__ inline v16bf load_A_f32(const float* base, int ld, int r16, int hi) {
  ABReg r;
  const float* p0 = base + (size_t)r16 * ld + hi * 8;
  const float* p1 = p0 + 16;
#pragma unroll
  for (int j = 0; j < 4; ++j) r.u[j]     = pack_bf16(p0[2*j], p0[2*j+1]);
#pragma unroll
  for (int j = 0; j < 4; ++j) r.u[4 + j] = pack_bf16(p1[2*j], p1[2*j+1]);
  return r.bf;
}
// B matrix 32x16 bf16, supplied as B^T row-major (16 rows of N, 32 cols of K):
// lane L holds column N=L&15; VGPR0-3: K = hi*8 + {0..7}; VGPR4-7: K = 16 + hi*8 + {0..7}.
static __device__ inline v16bf load_B_bf16(const u16* base, int ld, int r16, int hi) {
  ABReg r;
  const u16* p = base + (size_t)r16 * ld + hi * 8;
  r.q[0] = *(const uint4*)(p);        // K = hi*8 .. +7
  r.q[1] = *(const uint4*)(p + 16);   // K = 16+hi*8 .. +7
  return r.bf;
}

static __device__ inline v8f wmma_bf16(v16bf a, v16bf b, v8f c) {
  return __builtin_amdgcn_wmma_f32_16x16x32_bf16(false, a, false, b, (short)0, c, false, false);
}

// ---- CDNA5 async global->LDS copy (ASYNCcnt-tracked) ----
static __device__ inline void async_copy16(u32 lds_off, const void* gsrc) {
  asm volatile("global_load_async_to_lds_b128 %0, %1, off"
               :: "v"(lds_off), "v"(gsrc) : "memory");
}
template <int N>
static __device__ inline void wait_asynccnt() {
  asm volatile("s_wait_asynccnt %0" :: "i"(N) : "memory");
}

// Reductions across a 16-lane half (rows are striped across lanes 0-15 / 16-31)
static __device__ inline float half_reduce_max(float v) {
#pragma unroll
  for (int m = 1; m <= 8; m <<= 1) v = fmaxf(v, __shfl_xor(v, m, 32));
  return v;
}
static __device__ inline float half_reduce_sum(float v) {
#pragma unroll
  for (int m = 1; m <= 8; m <<= 1) v += __shfl_xor(v, m, 32);
  return v;
}

// ---------------- Kernel 0a: convert weights f32 -> bf16 ----------------
__global__ void cvt_w_kernel(const float* __restrict__ Wq, const float* __restrict__ Wk,
                             const float* __restrict__ Wv, u16* __restrict__ dst) {
  int i = blockIdx.x * blockDim.x + threadIdx.x;
  const int n = QD_ * D_;
  if (i < n) {
    dst[i]         = cvt_bf16(Wq[i]);
    dst[n + i]     = cvt_bf16(Wk[i]);
    dst[2 * n + i] = cvt_bf16(Wv[i]);
  }
}

// ---------------- Kernel 0b: convert x f32 -> bf16 (8 elems/thread) ----------------
__global__ void cvt_x_kernel(const float* __restrict__ x, u16* __restrict__ xb) {
  size_t i8 = ((size_t)blockIdx.x * blockDim.x + threadIdx.x) * 8;
  float4 a = *(const float4*)(x + i8);
  float4 b = *(const float4*)(x + i8 + 4);
  uint4 o;
  o.x = pack_bf16(a.x, a.y); o.y = pack_bf16(a.z, a.w);
  o.z = pack_bf16(b.x, b.y); o.w = pack_bf16(b.z, b.w);
  *(uint4*)(xb + i8) = o;
}

// ---------------- Kernel 1: QKV projection (bf16 WMMA, f32 accum) ----------------
// One wave computes one 16-row x 128-col tile of q, k, or v. A chunk k+1 is
// prefetched under the WMMA burst; B fragments are per-iteration.
template <bool XB>
__global__ void qkv_proj_kernel(const float* __restrict__ x, const u16* __restrict__ xb,
                                const u16* __restrict__ Wb,
                                u16* __restrict__ qb, u16* __restrict__ kb,
                                u16* __restrict__ vtb) {
  int wave = (blockIdx.x * blockDim.x + threadIdx.x) >> 5;
  int lane = threadIdx.x & 31;
  int r16 = lane & 15, hi = lane >> 4;
  int mat  = wave % 3;          // 0:q 1:k 2:v
  int tile = wave / 3;          // 0..BT/16-1
  int t0 = tile * 16;

  const u16* W = Wb + (size_t)mat * (QD_ * D_);
  v8f acc[8];
  v8f zero = {0.f, 0.f, 0.f, 0.f, 0.f, 0.f, 0.f, 0.f};
#pragma unroll
  for (int c = 0; c < 8; ++c) acc[c] = zero;

  auto loadA = [&](int d0) -> v16bf {
    if (XB) return load_A_bf16(xb + (size_t)t0 * D_ + d0, D_, r16, hi);
    else    return load_A_f32 (x  + (size_t)t0 * D_ + d0, D_, r16, hi);
  };

  v16bf A = loadA(0);
  for (int kk = 0; kk < D_ / 32; ++kk) {
    int d0 = kk * 32;
    v16bf Bf[8];
#pragma unroll
    for (int c = 0; c < 8; ++c)
      Bf[c] = load_B_bf16(W + (size_t)(c * 16) * D_ + d0, D_, r16, hi);
    v16bf An = A;
    if (kk + 1 < D_ / 32) An = loadA(d0 + 32);    // prefetch under the burst
#pragma unroll
    for (int c = 0; c < 8; ++c) acc[c] = wmma_bf16(A, Bf[c], acc[c]);
    A = An;
  }
  // D layout: lane holds column n = c*16 + r16; VGPR r -> row t0 + hi*8 + r
#pragma unroll
  for (int c = 0; c < 8; ++c) {
#pragma unroll
    for (int r = 0; r < 8; ++r) {
      int t  = t0 + hi * 8 + r;
      int nn = c * 16 + r16;
      u16 h = cvt_bf16(acc[c][r]);
      if (mat == 0)      qb[(size_t)t * QD_ + nn] = h;
      else if (mat == 1) kb[(size_t)t * QD_ + nn] = h;
      else {
        int bb = t / T_, sl = t - bb * T_;
        vtb[(size_t)bb * QD_ * T_ + (size_t)nn * T_ + sl] = h;
      }
    }
  }
}

// ---------------- Kernel 2: flash attention over keys ----------------
// Block = 4 waves = 4 query tiles of the SAME batch. K/V tiles for each 32-key
// block are staged once per workgroup in LDS via async global->LDS copies,
// double-buffered so stage i+1 overlaps the WMMAs of stage i.
// Q tiles live in LDS (per-wave) and ALL WMMA fragments are transient loads:
// persistent VGPRs ~= O(64)+m/l(16)+addresses -> no spills, high occupancy.
__global__ void attn_kernel(const u16* __restrict__ qb, const u16* __restrict__ kb,
                            const u16* __restrict__ vtb,
                            const unsigned char* __restrict__ mask,
                            float* __restrict__ out) {
  __shared__ u16 Kt[2][32 * QD_];     // 32 key rows x 128 dims   (8 KB each)
  __shared__ u16 Vt[2][QD_ * 32];     // 128 dims x 32 key cols   (8 KB each)
  __shared__ u16 Qt[4][16 * QD_];     // per-wave Q tile          (4 KB each)
  __shared__ u16 Plds[4][16 * 32];    // per-wave P transpose tile (1 KB each)

  int tid  = threadIdx.x;
  int w    = tid >> 5;
  int wave = blockIdx.x * 4 + w;
  int lane = tid & 31;
  int r16 = lane & 15, hi = lane >> 4;
  int t0 = wave * 16;                 // global (b*T + t) row
  int b  = t0 / T_;
  const u16* kbase = kb  + (size_t)b * T_ * QD_;
  const u16* vbase = vtb + (size_t)b * QD_ * T_;
  const unsigned char* mbase = mask + (size_t)b * T_;

  // Stage this wave's Q tile (16x128 bf16 = 4KB) into LDS once.
  {
    const u16* qg = qb + (size_t)t0 * QD_;
#pragma unroll
    for (int i = 0; i < 8; ++i) {
      int e = (lane + i * 32) * 8;                    // 8 u16 = 16B per lane
      *(uint4*)(&Qt[w][e]) = *(const uint4*)(qg + e);
    }
  }

  // Issue async copies for key-block starting at s0 into buffer bi (8 ops/thread).
  auto issue_tile = [&](int s0, int bi) {
    const u16* kg = kbase + (size_t)s0 * QD_;         // 8KB contiguous
    u32 koff = (u32)(size_t)(&Kt[bi][0]);
#pragma unroll
    for (int rpt = 0; rpt < 4; ++rpt) {
      int e = (tid + rpt * 128) * 8;                  // element offset (8 u16 = 16B)
      async_copy16(koff + e * 2, kg + e);
    }
    const u16* vg = vbase + (size_t)tid * T_ + s0;    // row tid of vT, 32 elems
    u32 voff = (u32)(size_t)(&Vt[bi][tid * 32]);
#pragma unroll
    for (int seg = 0; seg < 4; ++seg)
      async_copy16(voff + seg * 16, vg + seg * 8);
  };

  v8f zero = {0.f, 0.f, 0.f, 0.f, 0.f, 0.f, 0.f, 0.f};
  v8f O[8];
  float m[8], l[8];
#pragma unroll
  for (int c = 0; c < 8; ++c) O[c] = zero;
#pragma unroll
  for (int r = 0; r < 8; ++r) { m[r] = NEGV; l[r] = 0.f; }

  issue_tile(0, 0);                                   // prologue stage

  for (int kbi = 0; kbi < T_ / 32; ++kbi) {
    int s0 = kbi * 32;
    int bi = kbi & 1;
    if (kbi + 1 < T_ / 32) {
      issue_tile(s0 + 32, bi ^ 1);                    // prefetch next stage
      wait_asynccnt<8>();                             // current stage landed
    } else {
      wait_asynccnt<0>();
    }
    __syncthreads();                                  // cross-wave visibility

    // mask bytes early: latency hides under the S WMMAs
    float msk0 = mbase[s0 + r16]      ? 1.0f : 0.0f;
    float msk1 = mbase[s0 + 16 + r16] ? 1.0f : 0.0f;

    // ---- S = Q K^T : all fragments transient (Aq from LDS, reused twice) ----
    v8f S0 = zero, S1 = zero;
#pragma unroll
    for (int kk = 0; kk < 4; ++kk) {
      v16bf Aq = load_A_bf16(&Qt[w][kk * 32], QD_, r16, hi);
      v16bf B0 = load_B_bf16(&Kt[bi][0]        + kk * 32, QD_, r16, hi);
      v16bf B1 = load_B_bf16(&Kt[bi][16 * QD_] + kk * 32, QD_, r16, hi);
      S0 = wmma_bf16(Aq, B0, S0);
      S1 = wmma_bf16(Aq, B1, S1);
    }

    // ---- fused online softmax: all scalars, no live arrays ----
    u16* P = Plds[w];
#pragma unroll
    for (int r = 0; r < 8; ++r) {
      float a0 = msk0 > 0.f ? S0[r] : NEGV;
      float a1 = msk1 > 0.f ? S1[r] : NEGV;
      float rm = half_reduce_max(fmaxf(a0, a1));
      float mn = fmaxf(m[r], rm);
      float al = __expf(m[r] - mn);      // exp(<=0), no overflow
      m[r] = mn;
      float q0 = msk0 * __expf(a0 - mn); // masked -> 0 exactly
      float q1 = msk1 * __expf(a1 - mn);
      l[r] = l[r] * al + half_reduce_sum(q0 + q1);
      P[(hi * 8 + r) * 32 + r16]      = cvt_bf16(q0);
      P[(hi * 8 + r) * 32 + 16 + r16] = cvt_bf16(q1);
#pragma unroll
      for (int c = 0; c < 8; ++c) O[c][r] *= al;
    }

    __syncthreads();
    v16bf Ap = load_A_bf16(P, 32, r16, hi);

    // ---- O += P @ V (fragment groups of 4, transient) ----
    {
      v16bf Bv[4];
#pragma unroll
      for (int c = 0; c < 4; ++c)
        Bv[c] = load_B_bf16(&Vt[bi][(c * 16) * 32], 32, r16, hi);
#pragma unroll
      for (int c = 0; c < 4; ++c) O[c] = wmma_bf16(Ap, Bv[c], O[c]);
    }
    {
      v16bf Bv[4];
#pragma unroll
      for (int c = 0; c < 4; ++c)
        Bv[c] = load_B_bf16(&Vt[bi][((c + 4) * 16) * 32], 32, r16, hi);
#pragma unroll
      for (int c = 0; c < 4; ++c) O[c + 4] = wmma_bf16(Ap, Bv[c], O[c + 4]);
    }

    __syncthreads();                    // protect buffers before next prefetch/overwrite
  }

#pragma unroll
  for (int r = 0; r < 8; ++r) l[r] = (l[r] > 0.f) ? 1.0f / l[r] : 0.0f;
#pragma unroll
  for (int c = 0; c < 8; ++c)
#pragma unroll
    for (int r = 0; r < 8; ++r) {
      int t  = t0 + hi * 8 + r;
      int nn = c * 16 + r16;
      out[(size_t)t * QD_ + nn] = O[c][r] * l[r];
    }
}

extern "C" void kernel_launch(void* const* d_in, const int* in_sizes, int n_in,
                              void* d_out, int out_size, void* d_ws, size_t ws_size,
                              hipStream_t stream) {
  const float* x  = (const float*)d_in[0];
  const unsigned char* mask = (const unsigned char*)d_in[1];  // jnp.bool_ = 1 byte
  const float* Wq = (const float*)d_in[2];
  const float* Wk = (const float*)d_in[3];
  const float* Wv = (const float*)d_in[4];
  float* out = (float*)d_out;

  // Workspace (bf16): q[BT,128] | k[BT,128] | vT[B,128,T] | W[3,128,1024] | (opt) xb[BT,1024]
  u16* qb   = (u16*)d_ws;
  u16* kbuf = qb   + (size_t)BT_ * QD_;
  u16* vtb  = kbuf + (size_t)BT_ * QD_;
  u16* Wb   = vtb  + (size_t)BT_ * QD_;
  u16* xb   = Wb   + (size_t)3 * QD_ * D_;

  size_t need_base = ((size_t)3 * BT_ * QD_ + (size_t)3 * QD_ * D_) * 2;
  size_t need_xb   = need_base + (size_t)BT_ * D_ * 2;     // + 32 MB

  cvt_w_kernel<<<(QD_ * D_ + 255) / 256, 256, 0, stream>>>(Wq, Wk, Wv, Wb);
  if (ws_size >= need_xb) {
    cvt_x_kernel<<<((size_t)BT_ * D_ / 8 + 255) / 256, 256, 0, stream>>>(x, xb);
    qkv_proj_kernel<true ><<<(BT_ / 16 * 3) / 4, 128, 0, stream>>>(x, xb, Wb, qb, kbuf, vtb);
  } else {
    qkv_proj_kernel<false><<<(BT_ / 16 * 3) / 4, 128, 0, stream>>>(x, xb, Wb, qb, kbuf, vtb);
  }
  attn_kernel<<<(BT_ / 16) / 4, 128, 0, stream>>>(qb, kbuf, vtb, mask, out);
}
